// LSTM_81260781240588
// MI455X (gfx1250) — matrix operand order repaired and checked by reading, not compiled, not run on hardware
//
#include <hip/hip_runtime.h>
#include <hip/hip_bf16.h>

// ---------- types for WMMA ----------
typedef __attribute__((ext_vector_type(16))) __bf16 v16bf;
typedef __attribute__((ext_vector_type(8)))  float  v8f;

union FragBF { v16bf v; uint4 u[2]; };

#define HID   1024
#define XDIM  512
#define BATCH 64
#define TLEN  512
#define KTOT  1536            // HID + XDIM (fused [h|x] K dimension)
#define NTOT  4096            // 4 gates * HID

__device__ __forceinline__ unsigned short f32_to_bf16(float f) {
    unsigned int u = __float_as_uint(f);
    unsigned int lsb = (u >> 16) & 1u;
    u += 0x7fffu + lsb;                       // round to nearest even
    return (unsigned short)(u >> 16);
}

// Build fused, transposed, bf16 weight matrix: WcatT[n][k], n in [0,4096), k in [0,1536)
//   k <  1024 : W_h{gate}[k][col]     (gate = n>>10, col = n&1023)
//   k >= 1024 : W_x{gate}[k-1024][col]
__global__ __launch_bounds__(256) void prep_weights(
    const float* __restrict__ Whi, const float* __restrict__ Wxi,
    const float* __restrict__ Whf, const float* __restrict__ Wxf,
    const float* __restrict__ Who, const float* __restrict__ Wxo,
    const float* __restrict__ Whz, const float* __restrict__ Wxz,
    unsigned short* __restrict__ WcatT)
{
    int idx = blockIdx.x * blockDim.x + threadIdx.x;      // 0 .. NTOT*KTOT-1
    int n = idx / KTOT;
    int k = idx - n * KTOT;
    int gate = n >> 10;
    int col  = n & (HID - 1);
    const float* Wh[4] = {Whi, Whf, Who, Whz};
    const float* Wx[4] = {Wxi, Wxf, Wxo, Wxz};
    float w = (k < HID) ? Wh[gate][k * HID + col]
                        : Wx[gate][(k - HID) * HID + col];
    WcatT[idx] = f32_to_bf16(w);
}

// Init: h = 0, x-slot = bf16(X[:,0,:]), c_state = 0
__global__ __launch_bounds__(256) void init_state(
    const float* __restrict__ X,
    unsigned short* __restrict__ Abuf,   // [BATCH][KTOT] bf16: [h(1024) | x(512)]
    float* __restrict__ c_state)         // [BATCH][HID]
{
    int idx = blockIdx.x * blockDim.x + threadIdx.x;      // 0 .. BATCH*KTOT-1
    int b = idx / KTOT;
    int j = idx - b * KTOT;
    unsigned short v;
    if (j < HID) v = 0;                                   // h0 = 0
    else         v = f32_to_bf16(X[(b * TLEN + 0) * XDIM + (j - HID)]);
    Abuf[idx] = v;
    if (idx < BATCH * HID) c_state[idx] = 0.0f;
}

// One recurrent GEMM step: g[64][4096] = Abuf[64][1536] @ WcatT^T
// grid = 64 blocks (N chunks of 64), block = 128 threads (4 waves, one M-tile each)
__global__ __launch_bounds__(128) void gemm_step(
    const unsigned short* __restrict__ Abuf,
    const unsigned short* __restrict__ WcatT,
    float* __restrict__ g)
{
    const int lane  = threadIdx.x & 31;
    const int mtile = threadIdx.x >> 5;          // 0..3
    const int nbase = blockIdx.x * 64;           // 0..4032
    const int lm    = lane & 15;
    const int ha    = (lane & 16) ? 8  : 0;      // A: K half-offset per lane group
    const int hb    = (lane & 16) ? 16 : 0;      // unused: B loads use ha, same as A

    const unsigned short* aRow = Abuf  + (mtile * 16 + lm) * KTOT;
    const unsigned short* b0   = WcatT + (nbase +  0 + lm) * KTOT;
    const unsigned short* b1   = WcatT + (nbase + 16 + lm) * KTOT;
    const unsigned short* b2   = WcatT + (nbase + 32 + lm) * KTOT;
    const unsigned short* b3   = WcatT + (nbase + 48 + lm) * KTOT;

    v8f acc0 = {}, acc1 = {}, acc2 = {}, acc3 = {};

    #pragma unroll 4
    for (int kc = 0; kc < KTOT; kc += 32) {
        FragBF a, f0, f1, f2, f3;
        // A fragment: lanes 0-15 -> K {kc..kc+7, kc+16..kc+23}; lanes 16-31 -> +8
        a.u[0] = *(const uint4*)(aRow + kc + ha);
        a.u[1] = *(const uint4*)(aRow + kc + 16 + ha);
        // B fragments: lanes 0-15 -> K {kc..kc+7, kc+16..kc+23}; lanes 16-31 -> +8
        f0.u[0] = *(const uint4*)(b0 + kc + ha); f0.u[1] = *(const uint4*)(b0 + kc + 16 + ha);
        f1.u[0] = *(const uint4*)(b1 + kc + ha); f1.u[1] = *(const uint4*)(b1 + kc + 16 + ha);
        f2.u[0] = *(const uint4*)(b2 + kc + ha); f2.u[1] = *(const uint4*)(b2 + kc + 16 + ha);
        f3.u[0] = *(const uint4*)(b3 + kc + ha); f3.u[1] = *(const uint4*)(b3 + kc + 16 + ha);

        acc0 = __builtin_amdgcn_wmma_f32_16x16x32_bf16(false, a.v, false, f0.v, (short)0, acc0, false, false);
        acc1 = __builtin_amdgcn_wmma_f32_16x16x32_bf16(false, a.v, false, f1.v, (short)0, acc1, false, false);
        acc2 = __builtin_amdgcn_wmma_f32_16x16x32_bf16(false, a.v, false, f2.v, (short)0, acc2, false, false);
        acc3 = __builtin_amdgcn_wmma_f32_16x16x32_bf16(false, a.v, false, f3.v, (short)0, acc3, false, false);
    }

    // C/D layout: VGPR v -> row (v + 8*(lane>=16)), lane&15 -> column
    const int rbase = mtile * 16 + ((lane & 16) ? 8 : 0);
    #pragma unroll
    for (int v = 0; v < 8; ++v) {
        float* gr = g + (rbase + v) * NTOT + nbase + lm;
        gr[0]  = acc0[v];
        gr[16] = acc1[v];
        gr[32] = acc2[v];
        gr[48] = acc3[v];
    }
}

// Elementwise gate math + state update + staging of next-step activations
__global__ __launch_bounds__(256) void lstm_point(
    const float* __restrict__ g,
    const float* __restrict__ b_i, const float* __restrict__ b_f,
    const float* __restrict__ b_o, const float* __restrict__ b_z,
    float* __restrict__ c_state,
    unsigned short* __restrict__ Abuf,
    const float* __restrict__ X,
    float* __restrict__ out, int t)
{
    int idx = blockIdx.x * blockDim.x + threadIdx.x;      // 0 .. BATCH*HID-1
    int b = idx >> 10;
    int h = idx & (HID - 1);
    const float* gb = g + b * NTOT;

    float gi = gb[h]            + b_i[h];
    float gf = gb[HID + h]      + b_f[h];
    float go = gb[2 * HID + h]  + b_o[h];
    float gz = gb[3 * HID + h]  + b_z[h];

    float i_ = 1.0f / (1.0f + __expf(-gi));
    float f_ = 1.0f / (1.0f + __expf(-gf));
    float o_ = 1.0f / (1.0f + __expf(-go));
    float z_ = tanhf(gz);

    float c = i_ * z_ + f_ * c_state[idx];
    c_state[idx] = c;
    float hv = o_ * tanhf(c);

    out[(b * TLEN + t) * HID + h] = hv;
    Abuf[b * KTOT + h] = f32_to_bf16(hv);                 // h for next step (bf16)
    if (h < XDIM && (t + 1) < TLEN)                       // stage x_{t+1} (bf16)
        Abuf[b * KTOT + HID + h] =
            f32_to_bf16(X[(b * TLEN + (t + 1)) * XDIM + h]);
}

extern "C" void kernel_launch(void* const* d_in, const int* in_sizes, int n_in,
                              void* d_out, int out_size, void* d_ws, size_t ws_size,
                              hipStream_t stream) {
    const float* X   = (const float*)d_in[0];
    const float* Whi = (const float*)d_in[1];
    const float* Wxi = (const float*)d_in[2];
    const float* b_i = (const float*)d_in[3];
    const float* Whf = (const float*)d_in[4];
    const float* Wxf = (const float*)d_in[5];
    const float* b_f = (const float*)d_in[6];
    const float* Who = (const float*)d_in[7];
    const float* Wxo = (const float*)d_in[8];
    const float* b_o = (const float*)d_in[9];
    const float* Whz = (const float*)d_in[10];
    const float* Wxz = (const float*)d_in[11];
    const float* b_z = (const float*)d_in[12];

    // ---- workspace carve (all 256B-aligned) ----
    char* ws = (char*)d_ws;
    const size_t WCAT_B = (size_t)NTOT * KTOT * 2;        // 12,582,912
    const size_t ABUF_B = (size_t)BATCH * KTOT * 2;       //    196,608
    const size_t G_B    = (size_t)BATCH * NTOT * 4;       //  1,048,576
    unsigned short* WcatT = (unsigned short*)ws;
    unsigned short* Abuf  = (unsigned short*)(ws + WCAT_B);
    float*          g     = (float*)(ws + WCAT_B + ABUF_B);
    float*          cst   = (float*)(ws + WCAT_B + ABUF_B + G_B);

    // 1) fuse + transpose + bf16-convert weights (stays resident in L2 across steps)
    prep_weights<<<(NTOT * KTOT) / 256, 256, 0, stream>>>(
        Whi, Wxi, Whf, Wxf, Who, Wxo, Whz, Wxz, WcatT);

    // 2) init h=0, c=0, stage x_0
    init_state<<<(BATCH * KTOT) / 256, 256, 0, stream>>>(X, Abuf, cst);

    // 3) sequential scan: per step one WMMA GEMM + one gate kernel
    for (int t = 0; t < TLEN; ++t) {
        gemm_step<<<NTOT / 64, 128, 0, stream>>>(Abuf, WcatT, g);
        lstm_point<<<(BATCH * HID) / 256, 256, 0, stream>>>(
            g, b_i, b_f, b_o, b_z, cst, Abuf, X, (float*)d_out, t);
    }
}
